// OneModel_26499948216343
// MI455X (gfx1250) — matrix operand order repaired and checked by reading, not compiled, not run on hardware
//
#include <hip/hip_runtime.h>

// ---------------------------------------------------------------------------
// Problem constants (from reference): B=4, C=256, H=W=60, N=3600, out=[B,265,60,60]
// ---------------------------------------------------------------------------
#define B_    4
#define C_    256
#define HH_   60
#define WW_   60
#define HW_   3600
#define OUTC_ 265
#define EPSF  1e-7f
#define NEG_HUGE -3.402823e38f

typedef __bf16 bf16_t;
typedef __attribute__((ext_vector_type(16))) __bf16 v16bf;
typedef __attribute__((ext_vector_type(8)))  __bf16 v8bf;
typedef __attribute__((ext_vector_type(8)))  float  v8f;

// ---------------------------------------------------------------------------
// WMMA fragment helpers (CDNA5 wave32 layouts, 16-bit operands)
// A 16x32 (MxK): lane<16 -> row M=lane, K={k0..k0+7, k0+16..k0+23}
//                lane>=16 -> row M=lane-16, K={k0+8..k0+15, k0+24..k0+31}
// B 32x16 (KxN): lane<16 -> col N=lane, K={k0..k0+7, k0+16..k0+23}
//                lane>=16 -> col N=lane-16, K={k0+8..k0+15, k0+24..k0+31}
// C/D 16x16 f32: col = lane&15, row = v + 8*(lane>>4)
// ---------------------------------------------------------------------------
__device__ __forceinline__ v16bf frag_a16(const bf16_t* base, int ld, int row0, int k0, int lane) {
  const int hi = lane >> 4, r = lane & 15;
  const bf16_t* p = base + (size_t)(row0 + r) * ld + k0 + hi * 8;
  union { v16bf v; v8bf h[2]; } u;
  u.h[0] = *(const v8bf*)(p);
  u.h[1] = *(const v8bf*)(p + 16);
  return u.v;
}

__device__ __forceinline__ v16bf frag_b16(const bf16_t* base, int ld, int col0, int k0, int lane) {
  const int hi = lane >> 4, c = lane & 15;
  const bf16_t* p = base + (size_t)(col0 + c) * ld + k0 + hi * 8;
  union { v16bf v; v8bf h[2]; } u;
  u.h[0] = *(const v8bf*)(p);
  u.h[1] = *(const v8bf*)(p + 16);
  return u.v;
}

__device__ __forceinline__ v8f wmma_bf16(v16bf a, v16bf b, v8f c) {
  return __builtin_amdgcn_wmma_f32_16x16x32_bf16(false, a, false, b, (short)0, c, false, false);
}

// ---------------------------------------------------------------------------
// 1) prep: norms + bf16 operand panels
//    q_nc  [B][N][C] raw query rows        (attn A operand)
//    kf_nc [B][N][C] masked support rows   (attn B operand)
//    qhat  [B][N][C] normalized query rows (corr B op, sim A&B op)
//    shat  [B][N][C] normalized supp rows  (A1 precompute)
//    s_cn  [B][C][N] raw support           (PV B operand)
// ---------------------------------------------------------------------------
__global__ void prep_kernel(const float* __restrict__ q, const float* __restrict__ s,
                            const float* __restrict__ prior,
                            bf16_t* __restrict__ q_nc, bf16_t* __restrict__ kf_nc,
                            bf16_t* __restrict__ qhat, bf16_t* __restrict__ shat,
                            bf16_t* __restrict__ s_cn,
                            float* __restrict__ qnorm, float* __restrict__ tsnorm) {
  int i = blockIdx.x * blockDim.x + threadIdx.x;
  if (i >= B_ * HW_) return;
  int b = i / HW_, n = i % HW_;
  const float* qb = q + (size_t)b * C_ * HW_;
  const float* sb = s + (size_t)b * C_ * HW_;
  float sq = 0.f, ss = 0.f;
  for (int c = 0; c < C_; ++c) {
    float qv = qb[(size_t)c * HW_ + n];
    float sv = sb[(size_t)c * HW_ + n];
    sq += qv * qv; ss += sv * sv;
  }
  float qn = sqrtf(sq), sn = sqrtf(ss);
  float rq = 1.0f / (qn + 1e-30f);
  float rs = 1.0f / (sn + 1e-30f);
  float pm  = prior[(size_t)b * HW_ + n];
  float bin = pm > 0.5f ? 1.f : 0.f;
  qnorm[i]  = qn;
  tsnorm[i] = pm * qn;
  bf16_t* qr = q_nc + (size_t)i * C_;
  bf16_t* kr = kf_nc + (size_t)i * C_;
  bf16_t* qh = qhat + (size_t)i * C_;
  bf16_t* sh = shat + (size_t)i * C_;
  bf16_t* sc = s_cn + (size_t)b * C_ * HW_ + n;
  for (int c = 0; c < C_; ++c) {
    float qv = qb[(size_t)c * HW_ + n];
    float sv = sb[(size_t)c * HW_ + n];
    qr[c] = (bf16_t)qv;
    kr[c] = (bf16_t)(sv * bin);
    qh[c] = (bf16_t)(qv * rq);
    sh[c] = (bf16_t)(sv * rs);
    sc[(size_t)c * HW_] = (bf16_t)sv;
  }
}

// ---------------------------------------------------------------------------
// 2) A1[b][o][c] = sum_n W1[o][n] * shat[b][n][c]   (32x256 per batch, tiny)
// ---------------------------------------------------------------------------
__global__ void a1_kernel(const float* __restrict__ W1, const bf16_t* __restrict__ shat,
                          bf16_t* __restrict__ A1) {
  int i = blockIdx.x * blockDim.x + threadIdx.x;
  if (i >= B_ * 32 * C_) return;
  int b = i / (32 * C_);
  int o = (i / C_) % 32;
  int c = i % C_;
  const bf16_t* sb = shat + (size_t)b * HW_ * C_ + c;
  const float*  w  = W1 + (size_t)o * HW_;
  float acc = 0.f;
  for (int n = 0; n < HW_; ++n)
    acc += w[n] * (float)sb[(size_t)n * C_];
  A1[i] = (bf16_t)acc;
}

// ---------------------------------------------------------------------------
// 3) x1[b][32][N] = relu(A1 [32xC] x qhat^T [CxN])  — WMMA GEMM, 1 wave/tile
// ---------------------------------------------------------------------------
__global__ void x1_gemm_kernel(const bf16_t* __restrict__ A1, const bf16_t* __restrict__ qhat,
                               float* __restrict__ x1) {
  int lane = threadIdx.x;
  int blk  = blockIdx.x;
  int b  = blk / (HW_ / 16);
  int m0 = (blk % (HW_ / 16)) * 16;
  const bf16_t* Ab = A1 + (size_t)b * 32 * C_;
  const bf16_t* Qb = qhat + (size_t)b * HW_ * C_;
  v8f acc0 = {}, acc1 = {};
#pragma unroll
  for (int k = 0; k < C_; k += 32) {
    v16bf a0 = frag_a16(Ab, C_, 0, k, lane);
    v16bf a1 = frag_a16(Ab, C_, 16, k, lane);
    v16bf bb = frag_b16(Qb, C_, m0, k, lane);
    acc0 = wmma_bf16(a0, bb, acc0);
    acc1 = wmma_bf16(a1, bb, acc1);
  }
  int m = m0 + (lane & 15);
  int hi = lane >> 4;
#pragma unroll
  for (int v = 0; v < 8; ++v) {
    int o = v + 8 * hi;
    x1[((size_t)b * 32 + o) * HW_ + m]      = fmaxf(acc0[v], 0.f);
    x1[((size_t)b * 32 + o + 16) * HW_ + m] = fmaxf(acc1[v], 0.f);
  }
}

// ---------------------------------------------------------------------------
// 4) direct 3x3 SAME conv + relu (small channel counts: VALU is fine)
// ---------------------------------------------------------------------------
__global__ void conv3x3_relu_kernel(const float* __restrict__ in, const float* __restrict__ w,
                                    float* __restrict__ out, int Cin, int Cout,
                                    int outBatchStride, int outChanOff) {
  int i = blockIdx.x * blockDim.x + threadIdx.x;
  if (i >= B_ * Cout * HW_) return;
  int b  = i / (Cout * HW_);
  int o  = (i / HW_) % Cout;
  int hw = i % HW_;
  int h = hw / WW_, x = hw % WW_;
  const float* ib = in + (size_t)b * Cin * HW_;
  float acc = 0.f;
  for (int ci = 0; ci < Cin; ++ci) {
    const float* wr = w + ((size_t)o * Cin + ci) * 9;
    const float* ir = ib + (size_t)ci * HW_;
#pragma unroll
    for (int ky = 0; ky < 3; ++ky) {
      int y = h + ky - 1;
      if (y < 0 || y >= HH_) continue;
#pragma unroll
      for (int kx = 0; kx < 3; ++kx) {
        int xx = x + kx - 1;
        if (xx < 0 || xx >= WW_) continue;
        acc += wr[ky * 3 + kx] * ir[y * WW_ + xx];
      }
    }
  }
  out[((size_t)b * outBatchStride + outChanOff + o) * HW_ + hw] = fmaxf(acc, 0.f);
}

// ---------------------------------------------------------------------------
// 5) attention pass 1: streamed QK^T via WMMA, online (max, sumexp, sum, cnt)
//    per row -> rowmax, rowinv(=1/sumexp), aw(=rowsum/(cnt+eps))
// ---------------------------------------------------------------------------
__global__ void attn_pass1_kernel(const bf16_t* __restrict__ q_nc, const bf16_t* __restrict__ kf_nc,
                                  float* __restrict__ rowmax, float* __restrict__ rowinv,
                                  float* __restrict__ aw) {
  int lane = threadIdx.x;
  int blk  = blockIdx.x;
  int b  = blk / (HW_ / 16);
  int n0 = (blk % (HW_ / 16)) * 16;
  const bf16_t* Qb = q_nc + (size_t)b * HW_ * C_;
  const bf16_t* Kb = kf_nc + (size_t)b * HW_ * C_;
  v16bf afr[8];
#pragma unroll
  for (int kk = 0; kk < 8; ++kk) afr[kk] = frag_a16(Qb, C_, n0, kk * 32, lane);
  float mx[8], se[8], sm[8], ct[8];
#pragma unroll
  for (int v = 0; v < 8; ++v) { mx[v] = NEG_HUGE; se[v] = 0.f; sm[v] = 0.f; ct[v] = 0.f; }
  for (int m0 = 0; m0 < HW_; m0 += 16) {
    v8f d = {};
#pragma unroll
    for (int kk = 0; kk < 8; ++kk) {
      v16bf bb = frag_b16(Kb, C_, m0, kk * 32, lane);
      d = wmma_bf16(afr[kk], bb, d);
    }
#pragma unroll
    for (int v = 0; v < 8; ++v) {
      float x = d[v] * 0.0625f;            // / sqrt(C)
      if (x != 0.f) {                      // masked cols produce exact 0
        float nm = fmaxf(mx[v], x);
        se[v] = se[v] * __expf(mx[v] - nm) + __expf(x - nm);
        mx[v] = nm;
        sm[v] += x;
        ct[v] += 1.f;
      }
    }
  }
  __shared__ float red[32][8][4];
#pragma unroll
  for (int v = 0; v < 8; ++v) {
    red[lane][v][0] = mx[v]; red[lane][v][1] = se[v];
    red[lane][v][2] = sm[v]; red[lane][v][3] = ct[v];
  }
  __syncthreads();
  if (lane < 16) {
    int r = lane, hi = r >> 3, v = r & 7, l0 = hi * 16;
    float gm = NEG_HUGE;
    for (int ls = l0; ls < l0 + 16; ++ls) gm = fmaxf(gm, red[ls][v][0]);
    float S = 0.f, sum = 0.f, cn = 0.f;
    for (int ls = l0; ls < l0 + 16; ++ls) {
      float lm = red[ls][v][0], lse = red[ls][v][1];
      if (lse > 0.f) S += lse * __expf(lm - gm);
      sum += red[ls][v][2]; cn += red[ls][v][3];
    }
    size_t idx = (size_t)b * HW_ + n0 + r;
    rowmax[idx] = gm;
    rowinv[idx] = (S > 0.f) ? 1.f / S : 0.f;
    aw[idx] = sum / (cn + EPSF);
  }
}

// ---------------------------------------------------------------------------
// 6) per-batch min/max reduction (for aw normalization)
// ---------------------------------------------------------------------------
__global__ void minmax_kernel(const float* __restrict__ v, float* __restrict__ bounds) {
  int b = blockIdx.x, t = threadIdx.x;
  __shared__ float smn[256], smx[256];
  float mn = 3.4e38f, mx = NEG_HUGE;
  for (int n = t; n < HW_; n += 256) {
    float x = v[(size_t)b * HW_ + n];
    mn = fminf(mn, x); mx = fmaxf(mx, x);
  }
  smn[t] = mn; smx[t] = mx; __syncthreads();
  for (int s = 128; s > 0; s >>= 1) {
    if (t < s) { smn[t] = fminf(smn[t], smn[t + s]); smx[t] = fmaxf(smx[t], smx[t + s]); }
    __syncthreads();
  }
  if (t == 0) { bounds[2 * b] = smn[0]; bounds[2 * b + 1] = smx[0]; }
}

// ---------------------------------------------------------------------------
// 7) attention pass 2: recompute tiles, P=softmax (bf16 via LDS rearrange),
//    PV WMMA, write aw_norm * (P V) into out channels [8..263]
//    grid covers (b, n-tile, c-half of 128) to bound VGPR pressure
// ---------------------------------------------------------------------------
__global__ void attn_pass2_kernel(const bf16_t* __restrict__ q_nc, const bf16_t* __restrict__ kf_nc,
                                  const bf16_t* __restrict__ s_cn,
                                  const float* __restrict__ rowmax, const float* __restrict__ rowinv,
                                  const float* __restrict__ aw, const float* __restrict__ awb,
                                  float* __restrict__ out) {
  int lane = threadIdx.x;
  int blk  = blockIdx.x;
  int b = blk / ((HW_ / 16) * 2);
  int r = blk % ((HW_ / 16) * 2);
  int n0 = (r >> 1) * 16;
  int c0 = (r & 1) * 128;
  const bf16_t* Qb = q_nc + (size_t)b * HW_ * C_;
  const bf16_t* Kb = kf_nc + (size_t)b * HW_ * C_;
  const bf16_t* Vb = s_cn + (size_t)b * C_ * HW_;
  int hi = lane >> 4, lc = lane & 15;
  v16bf afr[8];
#pragma unroll
  for (int kk = 0; kk < 8; ++kk) afr[kk] = frag_a16(Qb, C_, n0, kk * 32, lane);
  float rm[8], idn[8], an[8];
  float awmin = awb[2 * b];
  float awrng = 1.f / (awb[2 * b + 1] - awmin + EPSF);
#pragma unroll
  for (int v = 0; v < 8; ++v) {
    size_t idx = (size_t)b * HW_ + n0 + v + 8 * hi;
    rm[v] = rowmax[idx]; idn[v] = rowinv[idx];
    an[v] = (aw[idx] - awmin) * awrng;
  }
  v8f acc[8];
  v8f zero = {};
#pragma unroll
  for (int t = 0; t < 8; ++t) acc[t] = zero;
  __shared__ bf16_t Pl[16][32];
  for (int m0 = 0; m0 < HW_; m0 += 32) {
#pragma unroll
    for (int sub = 0; sub < 2; ++sub) {
      int m = m0 + sub * 16;
      if (m < HW_) {
        v8f d = {};
#pragma unroll
        for (int kk = 0; kk < 8; ++kk) {
          v16bf bb = frag_b16(Kb, C_, m, kk * 32, lane);
          d = wmma_bf16(afr[kk], bb, d);
        }
#pragma unroll
        for (int v = 0; v < 8; ++v) {
          float x = d[v] * 0.0625f;
          float p = (x != 0.f) ? __expf(x - rm[v]) * idn[v] : 0.f;
          Pl[v + 8 * hi][lc + sub * 16] = (bf16_t)p;
        }
      } else {
#pragma unroll
        for (int v = 0; v < 8; ++v) Pl[v + 8 * hi][lc + sub * 16] = (bf16_t)0.f;
      }
    }
    __syncthreads();
    v16bf pa = frag_a16(&Pl[0][0], 32, 0, 0, lane);
#pragma unroll
    for (int t = 0; t < 8; ++t) {
      v16bf bb = frag_b16(Vb, HW_, c0 + t * 16, m0, lane);
      acc[t] = wmma_bf16(pa, bb, acc[t]);
    }
    __syncthreads();
  }
#pragma unroll
  for (int t = 0; t < 8; ++t) {
    int c = c0 + t * 16 + lc;
    float* ob = out + ((size_t)b * OUTC_ + 8 + c) * HW_ + n0 + 8 * hi;
#pragma unroll
    for (int v = 0; v < 8; ++v) ob[v] = an[v] * acc[t][v];
  }
}

// ---------------------------------------------------------------------------
// 8) sim: streamed qhat x qhat^T, exact eps factor t/(t+eps), running col-max
// ---------------------------------------------------------------------------
__global__ void sim_kernel(const bf16_t* __restrict__ qhat, const float* __restrict__ qnorm,
                           const float* __restrict__ tsnorm, float* __restrict__ simmax) {
  int lane = threadIdx.x;
  int blk  = blockIdx.x;
  int b  = blk / (HW_ / 16);
  int m0 = (blk % (HW_ / 16)) * 16;
  const bf16_t* Qb = qhat + (size_t)b * HW_ * C_;
  int hi = lane >> 4, lc = lane & 15;
  v16bf bfr[8];
#pragma unroll
  for (int kk = 0; kk < 8; ++kk) bfr[kk] = frag_b16(Qb, C_, m0, kk * 32, lane);
  float qm = qnorm[(size_t)b * HW_ + m0 + lc];
  float best = NEG_HUGE;
  for (int r0 = 0; r0 < HW_; r0 += 16) {
    v8f d = {};
#pragma unroll
    for (int kk = 0; kk < 8; ++kk) {
      v16bf a = frag_a16(Qb, C_, r0, kk * 32, lane);
      d = wmma_bf16(a, bfr[kk], d);
    }
#pragma unroll
    for (int v = 0; v < 8; ++v) {
      int rr = r0 + v + 8 * hi;
      float t = tsnorm[(size_t)b * HW_ + rr] * qm;
      float s = d[v] * (t / (t + EPSF));
      best = fmaxf(best, s);
    }
  }
  float other = __shfl_xor(best, 16, 32);   // lane pair (l, l+16) shares column m
  best = fmaxf(best, other);
  if (lane < 16) simmax[(size_t)b * HW_ + m0 + lc] = best;
}

// ---------------------------------------------------------------------------
// 9) sim post: min/max normalize, write out channel 264
// ---------------------------------------------------------------------------
__global__ void simpost_kernel(const float* __restrict__ simmax, float* __restrict__ out) {
  int b = blockIdx.x, t = threadIdx.x;
  __shared__ float smn[256], smx[256];
  float mn = 3.4e38f, mx = NEG_HUGE;
  for (int n = t; n < HW_; n += 256) {
    float x = simmax[(size_t)b * HW_ + n];
    mn = fminf(mn, x); mx = fmaxf(mx, x);
  }
  smn[t] = mn; smx[t] = mx; __syncthreads();
  for (int s = 128; s > 0; s >>= 1) {
    if (t < s) { smn[t] = fminf(smn[t], smn[t + s]); smx[t] = fmaxf(smx[t], smx[t + s]); }
    __syncthreads();
  }
  float gmn = smn[0];
  float rng = 1.f / (smx[0] - smn[0] + EPSF);
  for (int n = t; n < HW_; n += 256)
    out[((size_t)b * OUTC_ + 264) * HW_ + n] = (simmax[(size_t)b * HW_ + n] - gmn) * rng;
}

// ---------------------------------------------------------------------------
// launch
// ---------------------------------------------------------------------------
extern "C" void kernel_launch(void* const* d_in, const int* in_sizes, int n_in,
                              void* d_out, int out_size, void* d_ws, size_t ws_size,
                              hipStream_t stream) {
  const float* q     = (const float*)d_in[0];
  const float* s     = (const float*)d_in[1];
  const float* prior = (const float*)d_in[2];
  const float* W1    = (const float*)d_in[3];
  const float* W2    = (const float*)d_in[4];
  const float* W3    = (const float*)d_in[5];
  const float* W4    = (const float*)d_in[6];
  float* out = (float*)d_out;

  char* w = (char*)d_ws;
  auto alloc = [&](size_t bytes) -> char* {
    char* p = w;
    w += (bytes + 255) & ~(size_t)255;
    return p;
  };
  const size_t panel = (size_t)B_ * HW_ * C_ * sizeof(bf16_t);   // 7.37 MB
  bf16_t* q_nc  = (bf16_t*)alloc(panel);
  bf16_t* kf_nc = (bf16_t*)alloc(panel);
  bf16_t* qhat  = (bf16_t*)alloc(panel);
  bf16_t* shat  = (bf16_t*)alloc(panel);
  bf16_t* s_cn  = (bf16_t*)alloc(panel + 4096);                  // slack: B-frag K overrun on last chunk
  bf16_t* A1    = (bf16_t*)alloc((size_t)B_ * 32 * C_ * sizeof(bf16_t));
  float* x1     = (float*)alloc((size_t)B_ * 32 * HW_ * 4);
  float* x2     = (float*)alloc((size_t)B_ * 16 * HW_ * 4);
  float* x3     = (float*)alloc((size_t)B_ * 8 * HW_ * 4);
  float* qnorm  = (float*)alloc((size_t)B_ * HW_ * 4);
  float* tsnorm = (float*)alloc((size_t)B_ * HW_ * 4);
  float* rowmax = (float*)alloc((size_t)B_ * HW_ * 4);
  float* rowinv = (float*)alloc((size_t)B_ * HW_ * 4);
  float* aw     = (float*)alloc((size_t)B_ * HW_ * 4);
  float* smx    = (float*)alloc((size_t)B_ * HW_ * 4);
  float* awb    = (float*)alloc(64);

  // 1) operand prep
  prep_kernel<<<(B_ * HW_ + 255) / 256, 256, 0, stream>>>(q, s, prior, q_nc, kf_nc, qhat,
                                                          shat, s_cn, qnorm, tsnorm);
  // 2) A1 = W1 * shat  (folds the N-dim correlation into a 32xC panel)
  a1_kernel<<<(B_ * 32 * C_ + 255) / 256, 256, 0, stream>>>(W1, shat, A1);
  // 3) x1 = relu(A1 * qhat^T)   [WMMA]
  x1_gemm_kernel<<<B_ * (HW_ / 16), 32, 0, stream>>>(A1, qhat, x1);
  // 4) conv chain 32->16->8->8 (conv4 writes out channels 0..7)
  conv3x3_relu_kernel<<<(B_ * 16 * HW_ + 255) / 256, 256, 0, stream>>>(x1, W2, x2, 32, 16, 16, 0);
  conv3x3_relu_kernel<<<(B_ * 8 * HW_ + 255) / 256, 256, 0, stream>>>(x2, W3, x3, 16, 8, 8, 0);
  conv3x3_relu_kernel<<<(B_ * 8 * HW_ + 255) / 256, 256, 0, stream>>>(x3, W4, out, 8, 8, OUTC_, 0);
  // 5) attention pass 1 (streamed WMMA + online softmax stats)
  attn_pass1_kernel<<<B_ * (HW_ / 16), 32, 0, stream>>>(q_nc, kf_nc, rowmax, rowinv, aw);
  // 6) aw min/max per batch
  minmax_kernel<<<B_, 256, 0, stream>>>(aw, awb);
  // 7) attention pass 2 (P build + PV WMMA, writes out channels 8..263)
  attn_pass2_kernel<<<B_ * (HW_ / 16) * 2, 32, 0, stream>>>(q_nc, kf_nc, s_cn, rowmax, rowinv,
                                                            aw, awb, out);
  // 8) sim column-max (streamed WMMA)
  sim_kernel<<<B_ * (HW_ / 16), 32, 0, stream>>>(qhat, qnorm, tsnorm, smx);
  // 9) sim min/max normalize -> out channel 264
  simpost_kernel<<<B_, 256, 0, stream>>>(smx, out);
}
